// Model_76879914598812
// MI455X (gfx1250) — matrix operand order repaired and checked by reading, not compiled, not run on hardware
//
#include <hip/hip_runtime.h>
#include <hip/hip_bf16.h>
#include <math.h>

// ---------------- problem constants ----------------
#define NAT   20000   // atoms
#define MNBR  20      // neighbors per type
#define BCRY  200     // crystals
#define HD    128     // hidden
#define ED    20      // rbf basis
#define EPAD  32      // rbf padded K for WMMA
#define TWOH  256     // 2*H
#define NCONV 3
#define PI_F  3.14159265358979323846f

typedef __attribute__((ext_vector_type(16))) __bf16 bf16x16;
typedef __attribute__((ext_vector_type(8)))  float  f32x8;

// ---------------- helpers ----------------
// branchless, overflow-safe: softplus(x) = max(x,0) + log1p(exp(-|x|))
__device__ __forceinline__ float softplusf(float x) {
    return fmaxf(x, 0.0f) + __logf(1.0f + __expf(-fabsf(x)));
}
__device__ __forceinline__ float sigmoidf(float x) {
    return 1.0f / (1.0f + __expf(-x));   // exp overflow -> 1/inf = 0, benign
}

// A fragment (16x32 bf16): lane = row (lane&15), K base = (lane<16?0:8).
// Holds K [kb..kb+7] then [kb+16..kb+23] -> two 16B contiguous loads.
__device__ __forceinline__ bf16x16 load_a_frag(const __bf16* p) {
    union { bf16x16 v; uint4 q[2]; } u;
    u.q[0] = *reinterpret_cast<const uint4*>(p);
    u.q[1] = *reinterpret_cast<const uint4*>(p + 16);
    return u.v;
}
// B fragment (32x16 bf16) from K-major (transposed) weights: lane = col
// (lane&15), K base = (lane<16?0:8); K [kb..kb+7] then [kb+16..kb+23].
__device__ __forceinline__ bf16x16 load_b_frag(const __bf16* p) {
    union { bf16x16 v; uint4 q[2]; } u;
    u.q[0] = *reinterpret_cast<const uint4*>(p);
    u.q[1] = *reinterpret_cast<const uint4*>(p + 16);
    return u.v;
}
__device__ __forceinline__ f32x8 wmma_bf16(bf16x16 a, bf16x16 b, f32x8 c) {
    return __builtin_amdgcn_wmma_f32_16x16x32_bf16(false, a, false, b,
                                                   (short)0, c, false, false);
}

// ---------------- kernels ----------------

// nodes = atoms_embed @ emb_W + emb_b    [N,13]x[13,128]
__global__ __launch_bounds__(256) void embed_kernel(
    const float* __restrict__ ae, const float* __restrict__ W,
    const float* __restrict__ b, float* __restrict__ nf,
    __bf16* __restrict__ nb)
{
    int id = blockIdx.x * 256 + threadIdx.x;     // over N*H
    int n = id >> 7, h = id & 127;
    float s = b[h];
    #pragma unroll
    for (int k = 0; k < 13; ++k) s += ae[n * 13 + k] * W[k * HD + h];
    nf[id] = s;
    nb[id] = (__bf16)s;
}

// rbf[edge, e] (bf16, K padded to 32)
__global__ __launch_bounds__(256) void rbf_kernel(
    const float* __restrict__ fea, float cutoff, __bf16* __restrict__ out)
{
    int id = blockIdx.x * 256 + threadIdx.x;     // over N*M*32
    int e = id & 31, edge = id >> 5;
    float d = fea[edge];
    float val = 0.0f;
    if (e < ED && d < cutoff) {
        float k = (float)(e + 1);
        float env = 0.5f * (__cosf(PI_F * d / cutoff) + 1.0f);
        val = __sinf(d * k * (PI_F / cutoff)) / d * env;
    }
    out[id] = (__bf16)val;
}

// Wt[col*128 + k] = gW[k*256 + col]  (bf16, K-major)   gW pre-offset to block row
__global__ __launch_bounds__(256) void convert_wt_kernel(
    const float* __restrict__ gW, __bf16* __restrict__ Wt)
{
    int id = blockIdx.x * 256 + threadIdx.x;     // 256*128
    int col = id >> 7, k = id & 127;
    Wt[col * HD + k] = (__bf16)gW[k * TWOH + col];
}

// Wc = fW @ gW_e  (folded edge weights, K-major, padded) and full bias
__global__ __launch_bounds__(256) void make_wc_kernel(
    const float* __restrict__ fW,  // [20,128]
    const float* __restrict__ fb,  // [128]
    const float* __restrict__ gWe, // gW rows 128..255, stride 256
    const float* __restrict__ gb,  // [256]
    __bf16* __restrict__ Wct,      // [256,32]
    float* __restrict__ biasf)     // [256]
{
    int col = threadIdx.x;
    for (int e = 0; e < EPAD; ++e) {
        float s = 0.0f;
        if (e < ED)
            for (int h = 0; h < HD; ++h)
                s += fW[e * HD + h] * gWe[h * TWOH + col];
        Wct[col * EPAD + e] = (__bf16)s;
    }
    float bs = gb[col];
    for (int h = 0; h < HD; ++h) bs += fb[h] * gWe[h * TWOH + col];
    biasf[col] = bs;
}

// out[N,256] = nodes_bf16[N,128] @ W(K-major bf16) (+bias). WMMA core.
__global__ __launch_bounds__(256) void node_gemm_kernel(
    const __bf16* __restrict__ A,   // [N,128]
    const __bf16* __restrict__ Wt,  // [256,128] K-major
    const float* __restrict__ bias, int useBias,
    float* __restrict__ out)        // [N,256]
{
    const int lane = threadIdx.x & 31;
    const int w    = threadIdx.x >> 5;       // 8 waves
    const int lo   = (lane < 16);
    const int c    = lane & 15;
    const int rowbase = blockIdx.x * 16;
    const int row  = rowbase + c;
    const int kA   = lo ? 0 : 8;
    const int col0 = (2 * w) * 16 + c;
    const int col1 = (2 * w + 1) * 16 + c;

    f32x8 acc0 = {}, acc1 = {};
    if (useBias) {
        float b0 = bias[col0], b1 = bias[col1];
        #pragma unroll
        for (int i = 0; i < 8; ++i) { acc0[i] = b0; acc1[i] = b1; }
    }
    #pragma unroll
    for (int k0 = 0; k0 < HD; k0 += 32) {
        bf16x16 Af = load_a_frag(A + row * HD + k0 + kA);
        bf16x16 B0 = load_b_frag(Wt + col0 * HD + k0 + (lo ? 0 : 8));
        bf16x16 B1 = load_b_frag(Wt + col1 * HD + k0 + (lo ? 0 : 8));
        acc0 = wmma_bf16(Af, B0, acc0);
        acc1 = wmma_bf16(Af, B1, acc1);
    }
    const int rs = lo ? 0 : 8;
    #pragma unroll
    for (int vg = 0; vg < 8; ++vg) {
        out[(rowbase + vg + rs) * TWOH + col0] = acc0[vg];
        out[(rowbase + vg + rs) * TWOH + col1] = acc1[vg];
    }
}

// Fused edge kernel: gated = rbf@Wc + S[n] + P[idx]; gate; reduce over M.
// Block = 16 nodes (320 edges). 8 waves <-> 8 paired col-tiles. WMMA core.
__global__ __launch_bounds__(256) void edge_kernel(
    const __bf16* __restrict__ rbf,   // [N*M, 32]
    const int*    __restrict__ idx,   // [N*M]
    const float*  __restrict__ selfS, // [N, 256]  (S + bias)
    const float*  __restrict__ Pc,    // [N, 256]
    const __bf16* __restrict__ Wct,   // [256, 32] K-major
    float*        __restrict__ outp)  // [N, 128]
{
    const int lane = threadIdx.x & 31;
    const int w    = threadIdx.x >> 5;
    const bool lo  = (lane < 16);
    const int c    = lane & 15;
    const int nodebase = blockIdx.x * 16;
    const int edgeBase = nodebase * MNBR;     // 320 edges per block

    // stage neighbor indices once per block (8x vmem reduction)
    __shared__ int s_idx[16 * MNBR];
    for (int i = threadIdx.x; i < 16 * MNBR; i += 256)
        s_idx[i] = idx[edgeBase + i];
    __syncthreads();

    const int colF = w * 16 + c;              // filter col 0..127
    const int colC = HD + colF;               // core col 128..255

    bf16x16 Bf = load_b_frag(Wct + colF * EPAD + (lo ? 0 : 8));
    bf16x16 Bc = load_b_frag(Wct + colC * EPAD + (lo ? 0 : 8));

    float sf[16], sc[16], acc[16];
    #pragma unroll
    for (int j = 0; j < 16; ++j) {
        sf[j] = selfS[(nodebase + j) * TWOH + colF];
        sc[j] = selfS[(nodebase + j) * TWOH + colC];
        acc[j] = 0.0f;
    }
    const int kA = lo ? 0 : 8;

    #pragma unroll
    for (int t = 0; t < 20; ++t) {
        const int rowe = edgeBase + t * 16 + c;
        bf16x16 Af = load_a_frag(rbf + rowe * EPAD + kA);
        f32x8 z = {};
        f32x8 Df = wmma_bf16(Af, Bf, z);
        f32x8 Dc = wmma_bf16(Af, Bc, z);
        #pragma unroll
        for (int vg = 0; vg < 8; ++vg) {
            const int rl = t * 16 + vg;       // row for lanes 0..15
            const int rh = rl + 8;            // row for lanes 16..31
            const int ii = s_idx[lo ? rl : rh];
            float f = Df[vg] + Pc[ii * TWOH + colF];
            float g = Dc[vg] + Pc[ii * TWOH + colC];
            const int jl = rl / MNBR, jh = rh / MNBR;   // compile-time
            f += lo ? sf[jl] : sf[jh];
            g += lo ? sc[jl] : sc[jh];
            float v = sigmoidf(f) * softplusf(g);
            if (jl == jh) acc[jl] += v;
            else { acc[jl] += lo ? v : 0.0f; acc[jh] += lo ? 0.0f : v; }
        }
    }
    #pragma unroll
    for (int j = 0; j < 16; ++j) {
        float v = acc[j] + __shfl_xor(acc[j], 16);
        if (lo) outp[(nodebase + j) * HD + colF] = v;
    }
}

// nodes = softplus(nodes + che + vdw); refresh bf16 mirror
__global__ __launch_bounds__(256) void combine_kernel(
    float* __restrict__ nf, __bf16* __restrict__ nb,
    const float* __restrict__ a, const float* __restrict__ b)
{
    int id = blockIdx.x * 256 + threadIdx.x;
    float v = softplusf(nf[id] + a[id] + b[id]);
    nf[id] = v;
    nb[id] = (__bf16)v;
}

// per-crystal mean pool (contiguous ranges from num_atoms prefix sum)
__global__ __launch_bounds__(128) void pool_kernel(
    const float* __restrict__ nodes, const int* __restrict__ na,
    float* __restrict__ crys)
{
    int b = blockIdx.x, c = threadIdx.x;
    __shared__ int s_start;
    if (c == 0) { int s = 0; for (int i = 0; i < b; ++i) s += na[i]; s_start = s; }
    __syncthreads();
    int start = s_start, cnt = na[b];
    float s = 0.0f;
    for (int a = 0; a < cnt; ++a) s += nodes[(start + a) * HD + c];
    crys[b * HD + c] = s / (float)cnt;
}

// head: softplus(softplus(crys)@fc1+b) @ outW + outb
__global__ __launch_bounds__(128) void head_kernel(
    const float* __restrict__ crys, const float* __restrict__ fc1W,
    const float* __restrict__ fc1b, const float* __restrict__ outW,
    const float* __restrict__ outb, float* __restrict__ out)
{
    int b = blockIdx.x, c = threadIdx.x;
    __shared__ float sp[HD];
    __shared__ float red[HD];
    sp[c] = softplusf(crys[b * HD + c]);
    __syncthreads();
    float s = fc1b[c];
    for (int k = 0; k < HD; ++k) s += sp[k] * fc1W[k * HD + c];
    s = softplusf(s);
    red[c] = s * outW[c];
    __syncthreads();
    for (int off = 64; off > 0; off >>= 1) {
        if (c < off) red[c] += red[c + off];
        __syncthreads();
    }
    if (c == 0) out[b] = red[0] + outb[0];
}

// ---------------- host launcher ----------------
extern "C" void kernel_launch(void* const* d_in, const int* in_sizes, int n_in,
                              void* d_out, int out_size, void* d_ws, size_t ws_size,
                              hipStream_t stream)
{
    const float* atoms_embed = (const float*)d_in[0];
    const float* che_fea     = (const float*)d_in[1];
    const float* vdw_fea     = (const float*)d_in[2];
    const int*   che_idx     = (const int*)d_in[3];
    const int*   vdw_idx     = (const int*)d_in[4];
    const int*   num_atoms   = (const int*)d_in[5];
    const float* emb_W       = (const float*)d_in[6];
    const float* emb_b       = (const float*)d_in[7];
    const float* che_fW      = (const float*)d_in[8];
    const float* che_fb      = (const float*)d_in[9];
    const float* che_gW      = (const float*)d_in[10];
    const float* che_gb      = (const float*)d_in[11];
    const float* vdw_fW      = (const float*)d_in[12];
    const float* vdw_fb      = (const float*)d_in[13];
    const float* vdw_gW      = (const float*)d_in[14];
    const float* vdw_gb      = (const float*)d_in[15];
    const float* fc1_W       = (const float*)d_in[16];
    const float* fc1_b       = (const float*)d_in[17];
    const float* out_W       = (const float*)d_in[18];
    const float* out_b       = (const float*)d_in[19];
    float* out = (float*)d_out;

    // workspace layout (256B aligned)
    char* ws = (char*)d_ws;
    size_t o = 0;
    auto take = [&](size_t bytes) { size_t r = o; o = (o + bytes + 255) & ~(size_t)255; return r; };
    float*  nodes_f32 = (float*) (ws + take((size_t)NAT * HD * 4));
    __bf16* nodes_b   = (__bf16*)(ws + take((size_t)NAT * HD * 2));
    __bf16* rbf_che   = (__bf16*)(ws + take((size_t)NAT * MNBR * EPAD * 2));
    __bf16* rbf_vdw   = (__bf16*)(ws + take((size_t)NAT * MNBR * EPAD * 2));
    float*  selfS     = (float*) (ws + take((size_t)NAT * TWOH * 4));
    float*  Pc        = (float*) (ws + take((size_t)NAT * TWOH * 4));
    float*  che_out   = (float*) (ws + take((size_t)NAT * HD * 4));
    float*  vdw_out   = (float*) (ws + take((size_t)NAT * HD * 4));
    __bf16* Wa_t      = (__bf16*)(ws + take((size_t)TWOH * HD * 2));
    __bf16* Wg_t      = (__bf16*)(ws + take((size_t)TWOH * HD * 2));
    __bf16* Wrbf_t    = (__bf16*)(ws + take((size_t)TWOH * EPAD * 2));
    float*  biasf     = (float*) (ws + take((size_t)TWOH * 4));
    float*  crys      = (float*) (ws + take((size_t)BCRY * HD * 4));

    const int gNH   = (NAT * HD) / 256;          // 10000
    const int gRBF  = (NAT * MNBR * EPAD) / 256; // 50000
    const int gTile = NAT / 16;                  // 1250

    embed_kernel<<<gNH, 256, 0, stream>>>(atoms_embed, emb_W, emb_b, nodes_f32, nodes_b);
    rbf_kernel<<<gRBF, 256, 0, stream>>>(che_fea, 8.0f,  rbf_che);
    rbf_kernel<<<gRBF, 256, 0, stream>>>(vdw_fea, 12.0f, rbf_vdw);

    for (int i = 0; i < NCONV; ++i) {
        const size_t gOff = (size_t)i * 3 * HD * TWOH;   // i*384*256
        // ---- che half ----
        convert_wt_kernel<<<128, 256, 0, stream>>>(che_gW + gOff,                 Wa_t);
        convert_wt_kernel<<<128, 256, 0, stream>>>(che_gW + gOff + 2 * HD * TWOH, Wg_t);
        make_wc_kernel<<<1, 256, 0, stream>>>(che_fW + (size_t)i * ED * HD,
                                              che_fb + (size_t)i * HD,
                                              che_gW + gOff + HD * TWOH,
                                              che_gb + (size_t)i * TWOH,
                                              Wrbf_t, biasf);
        node_gemm_kernel<<<gTile, 256, 0, stream>>>(nodes_b, Wa_t, biasf, 1, selfS);
        node_gemm_kernel<<<gTile, 256, 0, stream>>>(nodes_b, Wg_t, biasf, 0, Pc);
        edge_kernel<<<gTile, 256, 0, stream>>>(rbf_che, che_idx, selfS, Pc, Wrbf_t, che_out);
        // ---- vdw half ----
        convert_wt_kernel<<<128, 256, 0, stream>>>(vdw_gW + gOff,                 Wa_t);
        convert_wt_kernel<<<128, 256, 0, stream>>>(vdw_gW + gOff + 2 * HD * TWOH, Wg_t);
        make_wc_kernel<<<1, 256, 0, stream>>>(vdw_fW + (size_t)i * ED * HD,
                                              vdw_fb + (size_t)i * HD,
                                              vdw_gW + gOff + HD * TWOH,
                                              vdw_gb + (size_t)i * TWOH,
                                              Wrbf_t, biasf);
        node_gemm_kernel<<<gTile, 256, 0, stream>>>(nodes_b, Wa_t, biasf, 1, selfS);
        node_gemm_kernel<<<gTile, 256, 0, stream>>>(nodes_b, Wg_t, biasf, 0, Pc);
        edge_kernel<<<gTile, 256, 0, stream>>>(rbf_vdw, vdw_idx, selfS, Pc, Wrbf_t, vdw_out);
        // ---- residual + softplus ----
        combine_kernel<<<gNH, 256, 0, stream>>>(nodes_f32, nodes_b, che_out, vdw_out);
    }

    pool_kernel<<<BCRY, 128, 0, stream>>>(nodes_f32, num_atoms, crys);
    head_kernel<<<BCRY, 128, 0, stream>>>(crys, fc1_W, fc1_b, out_W, out_b, out);
}
